// EEGGraphTransformer_26250840113832
// MI455X (gfx1250) — matrix operand order repaired and checked by reading, not compiled, not run on hardware
//
#include <hip/hip_runtime.h>
#include <hip/hip_bf16.h>

// Problem constants (from reference)
static constexpr int Bsz = 1024;   // batch
static constexpr int Nn  = 64;     // nodes
static constexpr int Tt  = 2048;   // time
static constexpr int Dd  = 256;    // embed
static constexpr int Hh  = 8;      // heads
static constexpr int Cc  = 64;     // per-head dim
static constexpr int HC  = 512;    // H*C
static constexpr int Ee  = 1024;   // edges
static constexpr int OUTD = 6;
static constexpr int Mrows = Bsz * Nn;       // 65536
static constexpr int QKVS  = 4 * HC;         // 2048 fused output cols

typedef __attribute__((ext_vector_type(16))) __bf16 v16bf;
typedef __attribute__((ext_vector_type(8)))  __bf16 v8bf;
typedef __attribute__((ext_vector_type(4)))  __bf16 v4bf;
typedef __attribute__((ext_vector_type(8)))  float  v8f;

__device__ __forceinline__ __bf16 f2bf(float f) {
    union { float f; unsigned u; } c; c.f = f;
    unsigned r = c.u + 0x7FFFu + ((c.u >> 16) & 1u);   // RNE
    unsigned short h = (unsigned short)(r >> 16);
    return __builtin_bit_cast(__bf16, h);
}
__device__ __forceinline__ float bf2f(__bf16 b) {
    unsigned short h = __builtin_bit_cast(unsigned short, b);
    unsigned u = ((unsigned)h) << 16;
    return __builtin_bit_cast(float, u);
}

// A-fragment for 16x16x32 bf16: per lane, 16 contiguous K elems:
// lanes 0-15: K=[0..15]; lanes 16-31: K=[16..31]
__device__ __forceinline__ v16bf load_a_frag(const __bf16* row_base, int hs) {
    v8bf lo = *(const v8bf*)(row_base + hs * 16);       // 16B aligned
    v8bf hi = *(const v8bf*)(row_base + hs * 16 + 8);   // 16B aligned
    return __builtin_shufflevector(lo, hi, 0,1,2,3,4,5,6,7,8,9,10,11,12,13,14,15);
}

// CDNA5 async global->LDS copy (ISA 10 / 15.18.3, GV mode). LDS address is the
// low 32 bits of the generic shared pointer.
__device__ __forceinline__ void async_copy_b128(void* lds_dst, const void* gsrc) {
    unsigned lds_off = (unsigned)(unsigned long long)(uintptr_t)lds_dst;
    asm volatile("global_load_async_to_lds_b128 %0, %1, off"
                 :: "v"(lds_off), "v"(gsrc) : "memory");
}
__device__ __forceinline__ void wait_async() {
    asm volatile("s_wait_asynccnt 0x0" ::: "memory");
}

// ---------------------------------------------------------------- weight packing
// Fragment-major B layout: idx = (((ktile*NT)+ntile)*32 + lane)*16 + e
// holds W[ktile*32 + (lane>>4)*16 + e][ntile*16 + (lane&15)]  -> one 32B load/lane.
__global__ void pack_wp_frag_kernel(const float* __restrict__ Wp, __bf16* __restrict__ Wpk) {
    int idx = blockIdx.x * blockDim.x + threadIdx.x;
    if (idx >= Tt * Dd) return;
    constexpr int NT = Dd / 16;
    int e    = idx & 15;
    int lane = (idx >> 4) & 31;
    int tile = idx >> 9;
    int ktile = tile / NT, ntile = tile % NT;
    int k = ktile * 32 + (lane >> 4) * 16 + e;
    int n = ntile * 16 + (lane & 15);
    Wpk[idx] = f2bf(Wp[k * Dd + n]);
}

__global__ void pack_w2_frag_kernel(const float* __restrict__ Wq, const float* __restrict__ Wk,
                                    const float* __restrict__ Wv, const float* __restrict__ Ws,
                                    __bf16* __restrict__ Wpk) {
    int idx = blockIdx.x * blockDim.x + threadIdx.x;
    if (idx >= Dd * QKVS) return;
    constexpr int NT = QKVS / 16;
    int e    = idx & 15;
    int lane = (idx >> 4) & 31;
    int tile = idx >> 9;
    int ktile = tile / NT, ntile = tile % NT;
    int k = ktile * 32 + (lane >> 4) * 16 + e;
    int n = ntile * 16 + (lane & 15);
    int seg = n >> 9, col = n & 511;
    const float* W = (seg == 0) ? Wq : (seg == 1) ? Wk : (seg == 2) ? Wv : Ws;
    Wpk[idx] = f2bf(W[k * HC + col]);
}

__global__ void pack_b2_kernel(const float* __restrict__ bq, const float* __restrict__ bk,
                               const float* __restrict__ bv, const float* __restrict__ bs,
                               float* __restrict__ b2) {
    int j = blockIdx.x * blockDim.x + threadIdx.x;
    if (j >= QKVS) return;
    int seg = j >> 9, col = j & 511;
    const float* bb = (seg == 0) ? bq : (seg == 1) ? bk : (seg == 2) ? bv : bs;
    b2[j] = bb[col];
}

// ---------------------------------------------------------------- adjacency counts
__global__ void zero_adj_kernel(float* __restrict__ Adj) {
    int i = blockIdx.x * blockDim.x + threadIdx.x;
    if (i < Nn * Nn) Adj[i] = 0.0f;
}
__global__ void count_adj_kernel(const long long* __restrict__ EI, float* __restrict__ Adj) {
    int e = blockIdx.x * blockDim.x + threadIdx.x;
    if (e >= Ee) return;
    int src = (int)EI[e];
    int dst = (int)EI[Ee + e];
    atomicAdd(&Adj[dst * Nn + src], 1.0f);
}

// ---------------------------------------------------------------- GEMM1: Xp = X @ Wp + bp (bf16 out)
// grid: (Dd/64, Mrows/64). A: f32->bf16 via LDS. B: fragment-major from global (L2-resident).
__global__ __launch_bounds__(256) void gemm_xp_kernel(
    const float* __restrict__ X, const __bf16* __restrict__ Wpk,
    const float* __restrict__ bp, __bf16* __restrict__ Xpb)
{
    constexpr int ALD = 40;                  // bf16 elems per LDS row (80B, 16B-aligned)
    constexpr int NT  = Dd / 16;
    __shared__ __bf16 As[64 * ALD];
    const int tid = threadIdx.x;
    const int lane = tid & 31, wave = tid >> 5;
    const int wm = wave & 3, wn = wave >> 2;
    const int hs = lane >> 4, l15 = lane & 15;
    const int M0 = blockIdx.y * 64, N0 = blockIdx.x * 64;
    const int tn0 = (N0 >> 4) + wn * 2;

    v8f acc0 = {}; v8f acc1 = {};
    for (int kt = 0; kt < Tt / 32; ++kt) {
        // stage A: 64x32 f32 -> bf16, float4 loads, b64 LDS stores
        #pragma unroll
        for (int s = 0; s < 2; ++s) {
            int t4  = tid + s * 256;
            int row = t4 >> 3;
            int j4  = (t4 & 7) * 4;
            const float4 xv = *(const float4*)&X[(size_t)(M0 + row) * Tt + kt * 32 + j4];
            v4bf w; w[0] = f2bf(xv.x); w[1] = f2bf(xv.y); w[2] = f2bf(xv.z); w[3] = f2bf(xv.w);
            *(v4bf*)&As[row * ALD + j4] = w;
        }
        if (kt + 1 < Tt / 32)
            __builtin_prefetch(&X[(size_t)(M0 + (tid >> 3)) * Tt + (kt + 1) * 32 + (tid & 7) * 4], 0, 1);
        __syncthreads();

        v16bf a  = load_a_frag(&As[(wm * 16 + l15) * ALD], hs);
        v16bf b0 = *(const v16bf*)(Wpk + (((size_t)kt * NT + tn0 + 0) * 32 + lane) * 16);
        v16bf b1 = *(const v16bf*)(Wpk + (((size_t)kt * NT + tn0 + 1) * 32 + lane) * 16);
        acc0 = __builtin_amdgcn_wmma_f32_16x16x32_bf16(false, a, false, b0, (short)0, acc0, false, false);
        acc1 = __builtin_amdgcn_wmma_f32_16x16x32_bf16(false, a, false, b1, (short)0, acc1, false, false);
        __syncthreads();
    }
    // C/D layout: VGPR r -> M = r + 8*hs ; N = l15
    #pragma unroll
    for (int r = 0; r < 8; ++r) {
        int m  = M0 + wm * 16 + r + hs * 8;
        int c0 = N0 + wn * 32 + l15;
        int c1 = c0 + 16;
        Xpb[(size_t)m * Dd + c0] = f2bf(acc0[r] + bp[c0]);
        Xpb[(size_t)m * Dd + c1] = f2bf(acc1[r] + bp[c1]);
    }
}

// ---------------------------------------------------------------- GEMM2: Y = Xp @ [Wq|Wk|Wv|Ws] + b2
// grid: (QKVS/64, Mrows/64). A staged via async global->LDS DMA; B fragment-major from global.
__global__ __launch_bounds__(256) void gemm_qkvs_kernel(
    const __bf16* __restrict__ Xpb, const __bf16* __restrict__ Wpk,
    const float* __restrict__ b2, __bf16* __restrict__ Y)
{
    constexpr int ALD = 40;
    constexpr int NT  = QKVS / 16;
    __shared__ __bf16 As[64 * ALD];
    const int tid = threadIdx.x;
    const int lane = tid & 31, wave = tid >> 5;
    const int wm = wave & 3, wn = wave >> 2;
    const int hs = lane >> 4, l15 = lane & 15;
    const int M0 = blockIdx.y * 64, N0 = blockIdx.x * 64;
    const int tn0 = (N0 >> 4) + wn * 2;
    const int arow = tid >> 2, aj8 = (tid & 3) * 8;   // one b128 per thread

    v8f acc0 = {}; v8f acc1 = {};
    for (int kt = 0; kt < Dd / 32; ++kt) {
        async_copy_b128(&As[arow * ALD + aj8],
                        Xpb + (size_t)(M0 + arow) * Dd + kt * 32 + aj8);
        wait_async();
        __syncthreads();

        v16bf a  = load_a_frag(&As[(wm * 16 + l15) * ALD], hs);
        v16bf b0 = *(const v16bf*)(Wpk + (((size_t)kt * NT + tn0 + 0) * 32 + lane) * 16);
        v16bf b1 = *(const v16bf*)(Wpk + (((size_t)kt * NT + tn0 + 1) * 32 + lane) * 16);
        acc0 = __builtin_amdgcn_wmma_f32_16x16x32_bf16(false, a, false, b0, (short)0, acc0, false, false);
        acc1 = __builtin_amdgcn_wmma_f32_16x16x32_bf16(false, a, false, b1, (short)0, acc1, false, false);
        __syncthreads();
    }
    #pragma unroll
    for (int r = 0; r < 8; ++r) {
        int m  = M0 + wm * 16 + r + hs * 8;
        int c0 = N0 + wn * 32 + l15;
        int c1 = c0 + 16;
        Y[(size_t)m * QKVS + c0] = f2bf(acc0[r] + b2[c0]);
        Y[(size_t)m * QKVS + c1] = f2bf(acc1[r] + b2[c1]);
    }
}

// ---------------------------------------------------------------- attention per (b,h)
// S = q k^T /8 ; count-weighted masked softmax (duplicate edges => weights) ;
// agg = P V ; node_out = agg + skip ; column-mean into graph_rep.
__global__ __launch_bounds__(256) void attn_kernel(
    const __bf16* __restrict__ Y, const float* __restrict__ Adj,
    float* __restrict__ Grep)
{
    constexpr int QLD = 80;   // bf16 elems/row (160B, 32B-aligned rows)
    constexpr int SLD = 66;   // f32 stride
    __shared__ __bf16 qs[Nn * QLD];
    __shared__ __bf16 ks[Nn * QLD];
    __shared__ __bf16 vst[Cc * QLD];   // transposed: vst[c][src]
    __shared__ float  Sb[Nn * SLD];
    __shared__ float  den[Nn];
    __bf16* Pb = qs;          // reuse q buffer for P (q dead after S)
    float*  No = Sb;          // reuse S buffer for node_out

    const int b = blockIdx.x, h = blockIdx.y;
    const int tid = threadIdx.x;
    const int lane = tid & 31, wave = tid >> 5;
    const int tm = wave & 3, tnb = (wave >> 2) * 2;
    const int hs = lane >> 4, l15 = lane & 15;

    // stage q/k via async DMA; v loaded + transposed manually
    #pragma unroll
    for (int s = 0; s < 2; ++s) {
        int idx = tid + s * 256;
        int n = idx >> 3, j8 = (idx & 7) * 8;
        size_t base = (size_t)(b * Nn + n) * QKVS + h * Cc + j8;
        async_copy_b128(&qs[n * QLD + j8], Y + base + 0 * HC);
        async_copy_b128(&ks[n * QLD + j8], Y + base + 1 * HC);
        v8bf vv = *(const v8bf*)(Y + base + 2 * HC);
        #pragma unroll
        for (int e2 = 0; e2 < 8; ++e2) vst[(j8 + e2) * QLD + n] = vv[e2];
    }
    wait_async();
    __syncthreads();

    // ---- S = q @ k^T * 0.125
    {
        v8f acc0 = {}; v8f acc1 = {};
        #pragma unroll
        for (int kt = 0; kt < 2; ++kt) {
            v16bf a  = load_a_frag(&qs[(tm * 16 + l15) * QLD + kt * 32], hs);
            // B[k][src] = k[src][k]: contiguous 16 elems at K = kt*32 + hs*16
            v16bf b0 = *(const v16bf*)&ks[((tnb + 0) * 16 + l15) * QLD + kt * 32 + hs * 16];
            v16bf b1 = *(const v16bf*)&ks[((tnb + 1) * 16 + l15) * QLD + kt * 32 + hs * 16];
            acc0 = __builtin_amdgcn_wmma_f32_16x16x32_bf16(false, a, false, b0, (short)0, acc0, false, false);
            acc1 = __builtin_amdgcn_wmma_f32_16x16x32_bf16(false, a, false, b1, (short)0, acc1, false, false);
        }
        #pragma unroll
        for (int r = 0; r < 8; ++r) {
            int m = tm * 16 + r + hs * 8;
            Sb[m * SLD + (tnb + 0) * 16 + l15] = acc0[r] * 0.125f;
            Sb[m * SLD + (tnb + 1) * 16 + l15] = acc1[r] * 0.125f;
        }
    }
    __syncthreads();

    // ---- count-weighted masked softmax per dst row
    if (tid < Nn) {
        int r = tid;
        float m = -1e30f; bool any = false;
        for (int s = 0; s < Nn; ++s) {
            float cnt = Adj[r * Nn + s];
            if (cnt > 0.0f) { m = fmaxf(m, Sb[r * SLD + s]); any = true; }
        }
        if (!any) m = 0.0f;
        float d = 0.0f;
        for (int s = 0; s < Nn; ++s) {
            float cnt = Adj[r * Nn + s];
            float p = (cnt > 0.0f) ? cnt * __expf(Sb[r * SLD + s] - m) : 0.0f;
            d += p;
            Pb[r * QLD + s] = f2bf(p);
        }
        den[r] = d + 1e-16f;
    }
    __syncthreads();

    // ---- agg = P @ V ; node_out = agg/den + skip
    {
        v8f acc0 = {}; v8f acc1 = {};
        #pragma unroll
        for (int kt = 0; kt < 2; ++kt) {
            v16bf a  = load_a_frag(&Pb[(tm * 16 + l15) * QLD + kt * 32], hs);
            // B[src][c] = v[src][c] = vst[c][src]: contiguous 16 elems at src = kt*32 + hs*16
            v16bf b0 = *(const v16bf*)&vst[((tnb + 0) * 16 + l15) * QLD + kt * 32 + hs * 16];
            v16bf b1 = *(const v16bf*)&vst[((tnb + 1) * 16 + l15) * QLD + kt * 32 + hs * 16];
            acc0 = __builtin_amdgcn_wmma_f32_16x16x32_bf16(false, a, false, b0, (short)0, acc0, false, false);
            acc1 = __builtin_amdgcn_wmma_f32_16x16x32_bf16(false, a, false, b1, (short)0, acc1, false, false);
        }
        #pragma unroll
        for (int r = 0; r < 8; ++r) {
            int m = tm * 16 + r + hs * 8;
            float inv = 1.0f / den[m];
            int c0 = (tnb + 0) * 16 + l15;
            int c1 = (tnb + 1) * 16 + l15;
            size_t sbase = (size_t)(b * Nn + m) * QKVS + 3 * HC + h * Cc;
            No[m * SLD + c0] = acc0[r] * inv + bf2f(Y[sbase + c0]);
            No[m * SLD + c1] = acc1[r] * inv + bf2f(Y[sbase + c1]);
        }
    }
    __syncthreads();

    // ---- mean over nodes -> graph_rep[b, h*64 + c]
    if (tid < Cc) {
        int c = tid;
        float s = 0.0f;
        for (int m = 0; m < Nn; ++m) s += No[m * SLD + c];
        Grep[b * HC + h * Cc + c] = s * (1.0f / (float)Nn);
    }
}

// ---------------------------------------------------------------- final: out = graph_rep @ Wr + br
__global__ void final_proj_kernel(const float* __restrict__ Grep, const float* __restrict__ Wr,
                                  const float* __restrict__ br, float* __restrict__ out)
{
    int b = blockIdx.x, t = threadIdx.x;
    if (t < OUTD) {
        float acc = br[t];
        for (int j = 0; j < HC; ++j) acc += Grep[b * HC + j] * Wr[j * OUTD + t];
        out[b * OUTD + t] = acc;
    }
}

// ---------------------------------------------------------------- launcher
extern "C" void kernel_launch(void* const* d_in, const int* in_sizes, int n_in,
                              void* d_out, int out_size, void* d_ws, size_t ws_size,
                              hipStream_t stream)
{
    const float*     X  = (const float*)d_in[0];
    const long long* EI = (const long long*)d_in[1];   // int64 edge_index [2,E]
    const float*     Wp = (const float*)d_in[2];
    const float*     bp = (const float*)d_in[3];
    const float*     Wq = (const float*)d_in[4];
    const float*     bq = (const float*)d_in[5];
    const float*     Wk = (const float*)d_in[6];
    const float*     bk = (const float*)d_in[7];
    const float*     Wv = (const float*)d_in[8];
    const float*     bv = (const float*)d_in[9];
    const float*     Ws = (const float*)d_in[10];
    const float*     bs = (const float*)d_in[11];
    const float*     Wr = (const float*)d_in[12];
    const float*     br = (const float*)d_in[13];
    float* out = (float*)d_out;

    char* ws = (char*)d_ws;
    size_t off = 0;
    auto carve = [&](size_t bytes) {
        void* p = ws + off;
        off = (off + bytes + 255) & ~(size_t)255;
        return p;
    };
    __bf16* WpkP = (__bf16*)carve((size_t)Tt * Dd * 2);     // packed Wp fragments
    __bf16* WpkQ = (__bf16*)carve((size_t)Dd * QKVS * 2);   // packed fused qkvs fragments
    float*  b2   = (float*) carve((size_t)QKVS * 4);
    float*  Adj  = (float*) carve((size_t)Nn * Nn * 4);
    __bf16* Xpb  = (__bf16*)carve((size_t)Mrows * Dd * 2);
    __bf16* Yb   = (__bf16*)carve((size_t)Mrows * QKVS * 2);
    float*  Grep = (float*) carve((size_t)Bsz * HC * 4);
    (void)ws_size; (void)in_sizes; (void)n_in; (void)out_size;

    pack_wp_frag_kernel<<<(Tt * Dd + 255) / 256, 256, 0, stream>>>(Wp, WpkP);
    pack_w2_frag_kernel<<<(Dd * QKVS + 255) / 256, 256, 0, stream>>>(Wq, Wk, Wv, Ws, WpkQ);
    pack_b2_kernel     <<<(QKVS + 255) / 256, 256, 0, stream>>>(bq, bk, bv, bs, b2);
    zero_adj_kernel    <<<(Nn * Nn + 255) / 256, 256, 0, stream>>>(Adj);
    count_adj_kernel   <<<(Ee + 255) / 256, 256, 0, stream>>>(EI, Adj);

    gemm_xp_kernel  <<<dim3(Dd / 64,   Mrows / 64), 256, 0, stream>>>(X, WpkP, bp, Xpb);
    gemm_qkvs_kernel<<<dim3(QKVS / 64, Mrows / 64), 256, 0, stream>>>(Xpb, WpkQ, b2, Yb);
    attn_kernel     <<<dim3(Bsz, Hh), 256, 0, stream>>>(Yb, Adj, Grep);
    final_proj_kernel<<<Bsz, 32, 0, stream>>>(Grep, Wr, br, out);
}
